// GAThlayer_12635793785488
// MI455X (gfx1250) — compile-verified, matrix operand order checked by reading, not run on hardware
//
#include <hip/hip_runtime.h>
#include <math.h>

#define IN_F   256
#define HEADS  8
#define CH     32
#define OUT_F  256
#define NEG_SLOPE 0.2f
#define LN_EPS 1e-5f

typedef __attribute__((ext_vector_type(16))) __bf16 v16bf;
typedef __attribute__((ext_vector_type(8)))  __bf16 v8bf;
typedef __attribute__((ext_vector_type(8)))  float  v8f;

// ---------- atomics ----------
__device__ inline void atomicAddF32(float* p, float v) {
  __hip_atomic_fetch_add(p, v, __ATOMIC_RELAXED, __HIP_MEMORY_SCOPE_AGENT);
}
// float max via monotonic int/uint trick (buffer initialized to -inf)
__device__ inline void atomicMaxF32(float* p, float v) {
  if (v >= 0.0f) atomicMax((int*)p, __float_as_int(v));
  else           atomicMin((unsigned int*)p, __float_as_uint(v));
}

// ---------- prep: x -> bf16 ----------
__global__ void k_convert_x(const float* __restrict__ x, __bf16* __restrict__ xb, long long total) {
  long long i = (long long)blockIdx.x * blockDim.x + threadIdx.x;
  if (i < total) xb[i] = (__bf16)x[i];
}

// ---------- prep: W[k][n] -> WT[n][k] bf16 ----------
__global__ void k_transpose_w(const float* __restrict__ W, __bf16* __restrict__ WT) {
  int i = blockIdx.x * blockDim.x + threadIdx.x;     // IN_F*OUT_F = 65536
  int k = i / OUT_F, n = i % OUT_F;
  WT[(size_t)n * IN_F + k] = (__bf16)W[i];
}

// ---------- prep: zero out / init segment max & sum ----------
__global__ void k_init(float* __restrict__ out, float* __restrict__ segmax,
                       float* __restrict__ segsum, long long n_out, long long n_nh) {
  long long i = (long long)blockIdx.x * blockDim.x + threadIdx.x;
  if (i < n_out) out[i] = 0.0f;
  if (i < n_nh)  { segmax[i] = -INFINITY; segsum[i] = 0.0f; }
}

// ---------- GEMM: h = x @ W via v_wmma_f32_16x16x32_bf16 ----------
// One wave per 16x16 output tile. blockDim=256 (8 waves, wave32).
// grid = (ceil(n/16), OUT_F/16/8)
// Row index is CLAMPED (not predicated) so all fragment loads are
// unconditional -> no EXEC divergence in the WMMA loop; only stores guard n.
__global__ void k_gemm_wmma(const __bf16* __restrict__ Xb, const __bf16* __restrict__ WT,
                            float* __restrict__ H, int n) {
  const int wave = threadIdx.x >> 5;
  const int lane = threadIdx.x & 31;
  const int tm   = blockIdx.x;                 // M tile
  const int tn   = blockIdx.y * 8 + wave;      // N tile 0..15
  const int lm   = lane & 15;
  const int half = lane >> 4;                  // K-octet selector within fragment

  int row = tm * 16 + lm;                      // A: lane -> M row
  row = (row < n) ? row : (n - 1);             // clamp: loads always in-bounds
  const int col = tn * 16 + lm;                // B/D: lane -> N col

  const __bf16* arow = Xb + (size_t)row * IN_F + half * 8;
  const __bf16* brow = WT + (size_t)col * IN_F + half * 8;

  v8f acc = {};
#pragma unroll
  for (int k0 = 0; k0 < IN_F; k0 += 32) {
    // 16-bit A 16x32 layout: VGPR0-3 cover K 0-15 (halves offset by 8),
    // VGPR4-7 cover K 16-31.  B mirrored (lane -> column) from WT[n][k].
    v8bf alo = *(const v8bf*)(arow + k0);
    v8bf ahi = *(const v8bf*)(arow + k0 + 16);
    v8bf blo = *(const v8bf*)(brow + k0);
    v8bf bhi = *(const v8bf*)(brow + k0 + 16);
    v16bf a, b;
#pragma unroll
    for (int i = 0; i < 8; ++i) {
      a[i] = alo[i]; a[8 + i] = ahi[i];
      b[i] = blo[i]; b[8 + i] = bhi[i];
    }
    acc = __builtin_amdgcn_wmma_f32_16x16x32_bf16(
        /*neg_a=*/false, a, /*neg_b=*/false, b,
        /*c_mod=*/(short)0, acc, /*reuse_a=*/false, /*reuse_b=*/false);
  }
  // D 16x16 f32: lanes 0-15 hold M=v, lanes 16-31 hold M=8+v; N = lane&15
  const int m0 = tm * 16 + half * 8;
  float* hcol = H + (size_t)m0 * OUT_F + col;
  if (m0 + 7 < n) {                            // full-tile fast path
#pragma unroll
    for (int v = 0; v < 8; ++v) hcol[(size_t)v * OUT_F] = acc[v];
  } else {
#pragma unroll
    for (int v = 0; v < 8; ++v)
      if (m0 + v < n) hcol[(size_t)v * OUT_F] = acc[v];
  }
}

// ---------- attention coefficients: a_src/a_dst [n, H] ----------
// one block per node; wave w == head w (wave32)
__global__ void k_att(const float* __restrict__ H, const float* __restrict__ att_s,
                      const float* __restrict__ att_d, float* __restrict__ a_s,
                      float* __restrict__ a_d) {
  const int node = blockIdx.x;
  const int t = threadIdx.x;                   // 0..255  (= head*32 + c)
  float hv = H[(size_t)node * OUT_F + t];
  float vs = hv * att_s[t];
  float vd = hv * att_d[t];
#pragma unroll
  for (int off = 16; off > 0; off >>= 1) {
    vs += __shfl_xor(vs, off, 32);
    vd += __shfl_xor(vd, off, 32);
  }
  if ((t & 31) == 0) {
    int head = t >> 5;
    a_s[(size_t)node * HEADS + head] = vs;
    a_d[(size_t)node * HEADS + head] = vd;
  }
}

// ---------- edge logits + segment max ----------
__global__ void k_edge_logits(const long long* __restrict__ edges,
                              const float* __restrict__ a_s, const float* __restrict__ a_d,
                              float* __restrict__ ebuf, float* __restrict__ segmax,
                              long long E, int n) {
  long long idx = (long long)blockIdx.x * blockDim.x + threadIdx.x;
  long long tot = (E + n) * HEADS;
  if (idx >= tot) return;
  long long eid = idx >> 3;
  int head = (int)(idx & 7);
  long long s, d;
  if (eid < E) { s = edges[eid]; d = edges[E + eid]; }
  else         { s = d = eid - E; }            // self-loop
  float v = a_s[s * HEADS + head] + a_d[d * HEADS + head];
  v = (v > 0.0f) ? v : NEG_SLOPE * v;          // LeakyReLU(0.2)
  ebuf[idx] = v;
  atomicMaxF32(&segmax[d * HEADS + head], v);
}

// ---------- exp(e - max) + segment sum ----------
__global__ void k_edge_exp(const long long* __restrict__ edges,
                           float* __restrict__ ebuf, const float* __restrict__ segmax,
                           float* __restrict__ segsum, long long E, int n) {
  long long idx = (long long)blockIdx.x * blockDim.x + threadIdx.x;
  long long tot = (E + n) * HEADS;
  if (idx >= tot) return;
  long long eid = idx >> 3;
  int head = (int)(idx & 7);
  long long d = (eid < E) ? edges[E + eid] : (eid - E);
  float ex = __expf(ebuf[idx] - segmax[d * HEADS + head]);
  ebuf[idx] = ex;
  atomicAddF32(&segsum[d * HEADS + head], ex);
}

// ---------- message scatter: out[dst] += h[src] * alpha ----------
// one block (256 threads) per edge; thread = output channel
__global__ void k_scatter(const long long* __restrict__ edges,
                          const float* __restrict__ ebuf, const float* __restrict__ segsum,
                          const float* __restrict__ H, float* __restrict__ out,
                          long long E, int n) {
  long long eid = blockIdx.x;
  int c = threadIdx.x;                         // 0..255
  int head = c >> 5;
  long long s, d;
  if (eid < E) { s = edges[eid]; d = edges[E + eid]; }
  else         { s = d = eid - E; }
  float alpha = ebuf[eid * HEADS + head] / segsum[d * HEADS + head];
  float val = H[s * OUT_F + c] * alpha;
  atomicAddF32(&out[d * OUT_F + c], val);
}

// ---------- bias + LayerNorm + ReLU (in-place on out) ----------
__global__ void k_layernorm(float* __restrict__ out, const float* __restrict__ bias,
                            const float* __restrict__ gamma, const float* __restrict__ beta) {
  __shared__ float red[16];
  const int node = blockIdx.x;
  const int t = threadIdx.x;
  float v = out[(size_t)node * OUT_F + t] + bias[t];
  float s = v, s2 = v * v;
#pragma unroll
  for (int off = 16; off > 0; off >>= 1) {
    s  += __shfl_xor(s,  off, 32);
    s2 += __shfl_xor(s2, off, 32);
  }
  if ((t & 31) == 0) { red[t >> 5] = s; red[8 + (t >> 5)] = s2; }
  __syncthreads();
  float ts = 0.0f, ts2 = 0.0f;
#pragma unroll
  for (int i = 0; i < 8; ++i) { ts += red[i]; ts2 += red[8 + i]; }
  float mu  = ts * (1.0f / OUT_F);
  float var = ts2 * (1.0f / OUT_F) - mu * mu;
  float r = rsqrtf(var + LN_EPS);
  float o = (v - mu) * r * gamma[t] + beta[t];
  out[(size_t)node * OUT_F + t] = fmaxf(o, 0.0f);
}

extern "C" void kernel_launch(void* const* d_in, const int* in_sizes, int n_in,
                              void* d_out, int out_size, void* d_ws, size_t ws_size,
                              hipStream_t stream) {
  const float*     x     = (const float*)d_in[0];
  const long long* edges = (const long long*)d_in[1];   // int64 per reference
  const float*     W     = (const float*)d_in[2];
  const float*     att_s = (const float*)d_in[3];
  const float*     att_d = (const float*)d_in[4];
  const float*     bias  = (const float*)d_in[5];
  const float*     gamma = (const float*)d_in[6];
  const float*     beta  = (const float*)d_in[7];
  float* out = (float*)d_out;

  const int       n  = in_sizes[0] / IN_F;       // 50000
  const long long E  = in_sizes[1] / 2;          // 800000
  const long long Et = E + n;                    // with self-loops

  // workspace layout (256B aligned)
  char* base = (char*)d_ws;
  size_t off = 0;
  auto take = [&](size_t bytes) -> char* {
    char* p = base + off;
    off = (off + bytes + 255) & ~(size_t)255;
    return p;
  };
  __bf16* Xb    = (__bf16*)take((size_t)n * IN_F * sizeof(__bf16));
  __bf16* WT    = (__bf16*)take((size_t)IN_F * OUT_F * sizeof(__bf16));
  float*  Hbuf  = (float*) take((size_t)n * OUT_F * sizeof(float));
  float*  a_src = (float*) take((size_t)n * HEADS * sizeof(float));
  float*  a_dst = (float*) take((size_t)n * HEADS * sizeof(float));
  float*  smax  = (float*) take((size_t)n * HEADS * sizeof(float));
  float*  ssum  = (float*) take((size_t)n * HEADS * sizeof(float));
  float*  ebuf  = (float*) take((size_t)Et * HEADS * sizeof(float));
  (void)ws_size;

  const int B = 256;
  // prep
  long long nx = (long long)n * IN_F;
  k_convert_x<<<(int)((nx + B - 1) / B), B, 0, stream>>>(x, Xb, nx);
  k_transpose_w<<<(IN_F * OUT_F) / B, B, 0, stream>>>(W, WT);
  long long n_out = (long long)n * OUT_F, n_nh = (long long)n * HEADS;
  k_init<<<(int)((n_out + B - 1) / B), B, 0, stream>>>(out, smax, ssum, n_out, n_nh);

  // GEMM h = x @ W  (wave per 16x16 tile, 8 waves/block, 2 column halves)
  dim3 ggrid((n + 15) / 16, OUT_F / 16 / 8);
  k_gemm_wmma<<<ggrid, B, 0, stream>>>(Xb, WT, Hbuf, n);

  // attention coefficients
  k_att<<<n, B, 0, stream>>>(Hbuf, att_s, att_d, a_src, a_dst);

  // edge phase
  long long teh = Et * HEADS;
  int eblk = (int)((teh + B - 1) / B);
  k_edge_logits<<<eblk, B, 0, stream>>>(edges, a_src, a_dst, ebuf, smax, E, n);
  k_edge_exp<<<eblk, B, 0, stream>>>(edges, ebuf, smax, ssum, E, n);
  k_scatter<<<(int)Et, B, 0, stream>>>(edges, ebuf, ssum, Hbuf, out, E, n);

  // bias + LayerNorm + ReLU
  k_layernorm<<<n, B, 0, stream>>>(out, bias, gamma, beta);
}
